// CrossViewTransformer_11072425689280
// MI455X (gfx1250) — matrix operand order repaired and checked by reading, not compiled, not run on hardware
//
#include <hip/hip_runtime.h>
#include <hip/hip_bf16.h>

// ---------- types ----------
typedef __bf16 bf16_t;
typedef bf16_t bf16x8  __attribute__((ext_vector_type(8)));
typedef bf16_t bf16x16 __attribute__((ext_vector_type(16)));
typedef float  f32x8   __attribute__((ext_vector_type(8)));

#define WMMA_BF16 __builtin_amdgcn_wmma_f32_16x16x32_bf16

static __device__ __forceinline__ bf16x16 join8(bf16x8 lo, bf16x8 hi) {
  return __builtin_shufflevector(lo, hi, 0,1,2,3,4,5,6,7,8,9,10,11,12,13,14,15);
}

// problem constants
constexpr int BN   = 4;
constexpr int C    = 256;
constexpr int C8   = 32;
constexpr int HW   = 4096;   // 64*64
constexpr int KC   = 4608;   // 512*9

// ---------- conversion / packing kernels ----------
__global__ void cvt_f32_bf16_v4(const float4* __restrict__ src,
                                bf16_t* __restrict__ dst, int n4) {
  int i = blockIdx.x * blockDim.x + threadIdx.x;
  if (i < n4) {
    float4 f = src[i];
    bf16_t* d = dst + (size_t)i * 4;
    d[0] = (bf16_t)f.x; d[1] = (bf16_t)f.y; d[2] = (bf16_t)f.z; d[3] = (bf16_t)f.w;
  }
}

// front_x (B,C,HW) f32 -> cat[b][0:256][HW] bf16   (cat is (B,512,HW))
__global__ void cvt_frontx_cat(const float* __restrict__ src, bf16_t* __restrict__ cat) {
  int i = blockIdx.x * blockDim.x + threadIdx.x;     // over B*C*HW = 4<<20
  if (i < BN * C * HW) {
    int b   = i >> 20;                 // C*HW = 1<<20
    int rem = i & ((1 << 20) - 1);
    cat[(size_t)b * 512 * HW + rem] = (bf16_t)src[i];
  }
}

// Wf (O=256, I=512, 3,3) f32 -> wfb[o][p][i] bf16 with p = ky*3+kx  (K contiguous in i)
__global__ void repack_wf(const float* __restrict__ src, bf16_t* __restrict__ dst) {
  int i = blockIdx.x * blockDim.x + threadIdx.x;     // over 256*512*9
  if (i < 256 * 512 * 9) {
    int o   = i / (512 * 9);
    int rem = i % (512 * 9);
    int ic  = rem / 9;
    int p   = rem % 9;
    dst[(size_t)o * KC + p * 512 + ic] = (bf16_t)src[i];
  }
}

// ---------- q/k projection: out^T[j][32] = (W X)^T + bias ----------
__global__ __launch_bounds__(128) void gemm_qk(
    const bf16_t* __restrict__ wq, const bf16_t* __restrict__ wk,
    const bf16_t* __restrict__ xq,   // cross_x bf16 (B,256,HW)
    const bf16_t* __restrict__ cat,  // front_x lives in cat[b][0:256]
    const float* __restrict__ bq, const float* __restrict__ bk,
    bf16_t* __restrict__ qT, bf16_t* __restrict__ kT)
{
  const int b     = blockIdx.y;
  const int which = blockIdx.z;                 // 0 = k(front), 1 = q(cross)
  const bf16_t* Wm   = which ? wq : wk;
  const float*  bias = which ? bq : bk;
  const bf16_t* X    = which ? (xq  + (size_t)b * 256 * HW)
                             : (cat + (size_t)b * 512 * HW);
  bf16_t* outT = (which ? qT : kT) + (size_t)b * HW * C8;

  const int n0   = blockIdx.x * 64;
  const int tid  = threadIdx.x;
  const int wave = tid >> 5, lane = tid & 31;
  const int l15 = lane & 15, lhalf = lane & 16, asub = lhalf >> 1;

  __shared__ __align__(32) bf16_t Bt[64][32];
  f32x8 acc0 = {}, acc1 = {};

  for (int kc = 0; kc < 8; ++kc) {
    const int k0 = kc * 32;
    __syncthreads();
#pragma unroll
    for (int rep = 0; rep < 16; ++rep) {        // stage 64x32 B tile (transposed)
      int t = tid + rep * 128;
      int kk = t >> 6, nl = t & 63;
      Bt[nl][kk] = X[(size_t)(k0 + kk) * HW + n0 + nl];
    }
    __syncthreads();
    const bf16_t* Wr0 = Wm + (size_t)(l15) * C + k0;
    const bf16_t* Wr1 = Wm + (size_t)(16 + l15) * C + k0;
    bf16x16 a0 = join8(*(const bf16x8*)(Wr0 + asub), *(const bf16x8*)(Wr0 + 16 + asub));
    bf16x16 a1 = join8(*(const bf16x8*)(Wr1 + asub), *(const bf16x8*)(Wr1 + 16 + asub));
    bf16x16 bb = join8(*(const bf16x8*)&Bt[wave * 16 + l15][asub], *(const bf16x8*)&Bt[wave * 16 + l15][16 + asub]);
    acc0 = WMMA_BF16(false, a0, false, bb, (short)0, acc0, false, false);
    acc1 = WMMA_BF16(false, a1, false, bb, (short)0, acc1, false, false);
  }

  // write transposed: row j holds 32 contiguous channel values
  const int n = n0 + wave * 16 + l15;
  bf16_t* orow = outT + (size_t)n * C8;
#pragma unroll
  for (int rt = 0; rt < 2; ++rt) {
    f32x8 acc = rt ? acc1 : acc0;
    int mbase = rt * 16 + asub;
    bf16x8 ov;
#pragma unroll
    for (int r = 0; r < 8; ++r) ov[r] = (bf16_t)(acc[r] + bias[mbase + r]);
    *(bf16x8*)(orow + mbase) = ov;
  }
}

// ---------- fused energy = K^T Q with max/argmax over keys ----------
__global__ __launch_bounds__(128) void energy_max(
    const bf16_t* __restrict__ qT, const bf16_t* __restrict__ kT,
    float* __restrict__ S, int* __restrict__ argo)
{
  const int b = blockIdx.y;
  const int wave = threadIdx.x >> 5, lane = threadIdx.x & 31;
  const int l15 = lane & 15, lhalf = lane & 16, asub = lhalf >> 1;
  const int j = blockIdx.x * 64 + wave * 16 + l15;

  const bf16_t* qrow = qT + ((size_t)b * HW + j) * C8;
  bf16x16 bb = join8(*(const bf16x8*)(qrow + asub), *(const bf16x8*)(qrow + 16 + asub));         // queries = B matrix, loaded once
  const bf16_t* kbase = kT + (size_t)b * HW * C8;

  float best = -3.4e38f;
  int   barg = 0;
  for (int i0 = 0; i0 < HW; i0 += 16) {
    const bf16_t* krow = kbase + (size_t)(i0 + l15) * C8;
    bf16x16 a = join8(*(const bf16x8*)(krow + asub), *(const bf16x8*)(krow + 16 + asub));
    f32x8 c = {};
    c = WMMA_BF16(false, a, false, bb, (short)0, c, false, false);
#pragma unroll
    for (int r = 0; r < 8; ++r) {
      float v = c[r];
      int   ki = i0 + r + asub;                         // C-layout: rows r / r+8
      if (v > best) { best = v; barg = ki; }
    }
  }
  // merge the two lane-halves that hold the same query column
  float obest = __shfl_xor(best, 16, 32);
  int   oarg  = __shfl_xor(barg, 16, 32);
  if (obest > best || (obest == best && oarg < barg)) { best = obest; barg = oarg; }
  if (lane < 16) {
    S[(size_t)b * HW + j]    = best;
    argo[(size_t)b * HW + j] = barg;
  }
}

// ---------- v projection: v = Wv * xhat + bv  (bf16 out, (B,256,HW)) ----------
__global__ __launch_bounds__(256) void gemm_v(
    const bf16_t* __restrict__ wv, const bf16_t* __restrict__ xhat,
    const float* __restrict__ bv, bf16_t* __restrict__ vout)
{
  const int b  = blockIdx.y;
  const int n0 = blockIdx.x * 16;
  const int tid = threadIdx.x, wave = tid >> 5, lane = tid & 31;
  const int l15 = lane & 15, lhalf = lane & 16, asub = lhalf >> 1;
  const bf16_t* X = xhat + (size_t)b * C * HW;

  __shared__ __align__(32) bf16_t Bt[16][32];
  f32x8 acc0 = {}, acc1 = {};
  const int row0 = wave * 32;                    // 8 waves cover 256 rows

  for (int kc = 0; kc < 8; ++kc) {
    const int k0 = kc * 32;
    __syncthreads();
#pragma unroll
    for (int rep = 0; rep < 2; ++rep) {
      int t = tid + rep * 256;
      int kk = t >> 4, nl = t & 15;
      Bt[nl][kk] = X[(size_t)(k0 + kk) * HW + n0 + nl];
    }
    __syncthreads();
    const bf16_t* Wr0 = wv + (size_t)(row0 + l15) * C + k0;
    const bf16_t* Wr1 = wv + (size_t)(row0 + 16 + l15) * C + k0;
    bf16x16 a0 = join8(*(const bf16x8*)(Wr0 + asub), *(const bf16x8*)(Wr0 + 16 + asub));
    bf16x16 a1 = join8(*(const bf16x8*)(Wr1 + asub), *(const bf16x8*)(Wr1 + 16 + asub));
    bf16x16 bb = join8(*(const bf16x8*)&Bt[l15][asub], *(const bf16x8*)&Bt[l15][16 + asub]);
    acc0 = WMMA_BF16(false, a0, false, bb, (short)0, acc0, false, false);
    acc1 = WMMA_BF16(false, a1, false, bb, (short)0, acc1, false, false);
  }

  bf16_t* obase = vout + (size_t)b * C * HW;
  const int n = n0 + l15;
#pragma unroll
  for (int rt = 0; rt < 2; ++rt) {
    f32x8 acc = rt ? acc1 : acc0;
    int m0 = row0 + rt * 16 + asub;
#pragma unroll
    for (int r = 0; r < 8; ++r)
      obase[(size_t)(m0 + r) * HW + n] = (bf16_t)(acc[r] + bv[m0 + r]);
  }
}

// ---------- hard gather of best value per query into cat[:,256:512] ----------
__global__ void gather_cat(const bf16_t* __restrict__ v, const int* __restrict__ arg,
                           bf16_t* __restrict__ cat)
{
  int i = blockIdx.x * blockDim.x + threadIdx.x;   // over B*C*HW
  if (i >= BN * C * HW) return;
  int b = i >> 20;
  int c = (i >> 12) & 255;
  int j = i & 4095;
  int a = arg[(b << 12) + j];
  cat[((size_t)b * 512 + 256 + c) * HW + j] = v[((size_t)b * C + c) * HW + a];
}

// ---------- implicit-GEMM 3x3 conv + fused epilogue (N tile = 64) ----------
__global__ __launch_bounds__(256) void conv3x3_fuse(
    const bf16_t* __restrict__ wf,   // [256][9][512] bf16
    const bf16_t* __restrict__ cat,  // [B][512][HW] bf16
    const float* __restrict__ bfb, const float* __restrict__ S,
    const float* __restrict__ front_x, float* __restrict__ out)
{
  const int b  = blockIdx.y;
  const int n0 = blockIdx.x * 64;
  const int tid = threadIdx.x, wave = tid >> 5, lane = tid & 31;
  const int l15 = lane & 15, lhalf = lane & 16, asub = lhalf >> 1;
  const bf16_t* catb = cat + (size_t)b * 512 * HW;

  __shared__ __align__(32) bf16_t Bt[64][32];   // 4 KB, transposed [n][k]
  f32x8 acc[2][4] = {};                          // [row tile][col group]
  const int row0 = wave * 32;                    // 8 waves cover 256 output rows

  for (int p = 0; p < 9; ++p) {                  // conv tap
    const int dy = p / 3 - 1, dx = p % 3 - 1;
    for (int ic = 0; ic < 16; ++ic) {            // 512 input channels / 32
      const int i0 = ic * 32;
      __syncthreads();
#pragma unroll
      for (int rep = 0; rep < 8; ++rep) {        // stage shifted 64x32 B tile w/ zero pad
        int t = tid + rep * 256;
        int kk = t >> 6, nl = t & 63;
        int n = n0 + nl;
        int yy = (n >> 6) + dy, xx = (n & 63) + dx;
        bf16_t val = (bf16_t)0.0f;
        if ((unsigned)yy < 64u && (unsigned)xx < 64u)
          val = catb[(size_t)(i0 + kk) * HW + yy * 64 + xx];
        Bt[nl][kk] = val;
      }
      __syncthreads();
      const bf16_t* Wr0 = wf + (size_t)(row0 + l15) * KC + p * 512 + i0;
      const bf16_t* Wr1 = Wr0 + (size_t)16 * KC;
      __builtin_prefetch(Wr0 + 32, 0, 1);        // next channel chunk of weights
      bf16x16 a0 = join8(*(const bf16x8*)(Wr0 + asub), *(const bf16x8*)(Wr0 + 16 + asub));
      bf16x16 a1 = join8(*(const bf16x8*)(Wr1 + asub), *(const bf16x8*)(Wr1 + 16 + asub));
#pragma unroll
      for (int nc = 0; nc < 4; ++nc) {
        bf16x16 bb = join8(*(const bf16x8*)&Bt[nc * 16 + l15][asub], *(const bf16x8*)&Bt[nc * 16 + l15][16 + asub]);
        acc[0][nc] = WMMA_BF16(false, a0, false, bb, (short)0, acc[0][nc], false, false);
        acc[1][nc] = WMMA_BF16(false, a1, false, bb, (short)0, acc[1][nc], false, false);
      }
    }
  }

  const float* fx = front_x + (size_t)b * C * HW;
  float* ob = out + (size_t)b * C * HW;
#pragma unroll
  for (int nc = 0; nc < 4; ++nc) {
    const int n = n0 + nc * 16 + l15;
    const float s = S[(b << 12) + n];
#pragma unroll
    for (int rt = 0; rt < 2; ++rt) {
      int m0 = row0 + rt * 16 + asub;
#pragma unroll
      for (int r = 0; r < 8; ++r) {
        int m = m0 + r;
        size_t idx = (size_t)m * HW + n;
        ob[idx] = fx[idx] + (acc[rt][nc][r] + bfb[m]) * s;
      }
    }
  }
}

// ---------- launcher ----------
extern "C" void kernel_launch(void* const* d_in, const int* in_sizes, int n_in,
                              void* d_out, int out_size, void* d_ws, size_t ws_size,
                              hipStream_t stream)
{
  (void)in_sizes; (void)n_in; (void)out_size; (void)ws_size;
  const float* front_x = (const float*)d_in[0];
  const float* cross_x = (const float*)d_in[1];
  const float* xhat    = (const float*)d_in[2];
  const float* Wq  = (const float*)d_in[3];
  const float* bq  = (const float*)d_in[4];
  const float* Wk  = (const float*)d_in[5];
  const float* bk  = (const float*)d_in[6];
  const float* Wv  = (const float*)d_in[7];
  const float* bv  = (const float*)d_in[8];
  const float* Wf  = (const float*)d_in[9];
  const float* bfb = (const float*)d_in[10];
  float* out = (float*)d_out;

  // workspace carve-up (~46.7 MB total)
  char* ws = (char*)d_ws;
  size_t off = 0;
  auto carve = [&](size_t bytes) -> void* {
    void* p = ws + off;
    off = (off + bytes + 255) & ~(size_t)255;
    return p;
  };
  bf16_t* cat    = (bf16_t*)carve((size_t)BN * 512 * HW * 2);  // 16 MB
  bf16_t* crossb = (bf16_t*)carve((size_t)BN * C   * HW * 2);  // 8 MB
  bf16_t* xhatb  = (bf16_t*)carve((size_t)BN * C   * HW * 2);  // 8 MB
  bf16_t* wqb    = (bf16_t*)carve((size_t)C8 * C * 2);
  bf16_t* wkb    = (bf16_t*)carve((size_t)C8 * C * 2);
  bf16_t* wvb    = (bf16_t*)carve((size_t)C * C * 2);
  bf16_t* wfb    = (bf16_t*)carve((size_t)C * KC * 2);         // 2.36 MB
  bf16_t* qT     = (bf16_t*)carve((size_t)BN * HW * C8 * 2);   // 1 MB
  bf16_t* kT     = (bf16_t*)carve((size_t)BN * HW * C8 * 2);   // 1 MB
  bf16_t* vb     = (bf16_t*)carve((size_t)BN * C * HW * 2);    // 8 MB
  float*  Sb     = (float*)carve((size_t)BN * HW * 4);
  int*    argb   = (int*)carve((size_t)BN * HW * 4);

  const int nBig = BN * C * HW;       // 4 Mi
  dim3 blk(256);

  cvt_f32_bf16_v4<<<(nBig / 4 + 255) / 256, blk, 0, stream>>>((const float4*)cross_x, crossb, nBig / 4);
  cvt_f32_bf16_v4<<<(nBig / 4 + 255) / 256, blk, 0, stream>>>((const float4*)xhat, xhatb, nBig / 4);
  cvt_frontx_cat<<<(nBig + 255) / 256, blk, 0, stream>>>(front_x, cat);
  cvt_f32_bf16_v4<<<(C8 * C / 4 + 255) / 256, blk, 0, stream>>>((const float4*)Wq, wqb, C8 * C / 4);
  cvt_f32_bf16_v4<<<(C8 * C / 4 + 255) / 256, blk, 0, stream>>>((const float4*)Wk, wkb, C8 * C / 4);
  cvt_f32_bf16_v4<<<(C * C / 4 + 255) / 256, blk, 0, stream>>>((const float4*)Wv, wvb, C * C / 4);
  repack_wf<<<(C * KC + 255) / 256, blk, 0, stream>>>(Wf, wfb);

  gemm_qk<<<dim3(HW / 64, BN, 2), 128, 0, stream>>>(wqb, wkb, crossb, cat, bq, bk, qT, kT);
  energy_max<<<dim3(HW / 64, BN), 128, 0, stream>>>(qT, kT, Sb, argb);
  gemm_v<<<dim3(HW / 16, BN), 256, 0, stream>>>(wvb, xhatb, bv, vb);
  gather_cat<<<(nBig + 255) / 256, blk, 0, stream>>>(vb, argb, cat);
  conv3x3_fuse<<<dim3(HW / 64, BN), 256, 0, stream>>>(wfb, cat, bfb, Sb, front_x, out);
}
